// SwinMultiHeadSA_65257733095770
// MI455X (gfx1250) — matrix operand order repaired and checked by reading, not compiled, not run on hardware
//
#include <hip/hip_runtime.h>

// ---------------- types ----------------
typedef __bf16 bf16;
typedef __attribute__((ext_vector_type(16))) __bf16 v16bf;
typedef __attribute__((ext_vector_type(8)))  float  v8f;

#define D_DIM   256
#define HEADS   8
#define WSZ     7
#define P_TOK   49
#define PT      64          // padded tokens per window
#define CH      56          // patch grid side
#define N_BATCH 8
#define PN      3136
#define NWIN    64
#define WCNT    (2048*256)  // elements per weight matrix

// ws layout (bytes)
#define XS_BYTES   ((size_t)N_BATCH * PN * D_DIM * 2)   // 12,845,056
#define W_BYTES    ((size_t)WCNT * 2)                    // 1,048,576

// LDS row strides (padded for bank-conflict-free b128 fragment loads; 64 banks x 4B)
#define XS_S  264   // bf16, tokens x features   (528B row -> bank step 4, disjoint 4-bank groups)
#define VT_S  72    // bf16, features x tokens   (144B row -> bank step 36, gcd 4 -> disjoint)
#define AT_S  66    // f32,  attT[q][p]
#define WM_S  66    // bf16, wmT[q][p]           (132B row -> bank step 33, odd -> distinct)

// dynamic LDS layout (bytes)
#define L_XS   0                               // 64*264*2 = 33792
#define L_QS   33792                           // 33792
#define L_KS   67584                           // 33792 (reused as headT)
#define L_VT   101376                          // 256*72*2 = 36864
#define L_ATT  138240                          // 64*66*4 = 16896
#define L_WM   155136                          // 64*66*2 = 8448
#define L_TOT  163584                          // <= 160KB -> 2 workgroups/WGP

// ---------------- WMMA fragment loaders (ISA 7.12.2, wave32) ----------------
// A 16x32 bf16, source row-major [M][K] (K contiguous):
// lane&15 = M row; hi half-lanes K offset +8; VGPR i<4 -> K=2i, i>=4 -> K=2i+8
__device__ __forceinline__ v16bf load_a_rm(const bf16* base, int stride, int row0, int k0, int lane) {
  const bf16* p = base + (size_t)(row0 + (lane & 15)) * stride + k0 + ((lane >> 4) << 3);
  v16bf a;
#pragma unroll
  for (int i = 0; i < 8; ++i) {
    int k = (i < 4) ? 2 * i : 2 * i + 8;
    a[2 * i]     = p[k];
    a[2 * i + 1] = p[k + 1];
  }
  return a;
}
// B 32x16 bf16, source stored [N][K] (K contiguous):
// lane&15 = N col; hi half-lanes K offset +8; VGPR i<4 -> K=2i, i>=4 -> K=2i+8
__device__ __forceinline__ v16bf load_b_cm(const bf16* base, int stride, int k0, int col0, int lane) {
  const bf16* p = base + (size_t)(col0 + (lane & 15)) * stride + k0 + ((lane >> 4) << 3);
  v16bf b;
#pragma unroll
  for (int i = 0; i < 8; ++i) {
    int k = (i < 4) ? 2 * i : 2 * i + 8;
    b[2 * i]     = p[k];
    b[2 * i + 1] = p[k + 1];
  }
  return b;
}

__device__ __forceinline__ v8f v8f_zero() {
  v8f z = {0.f, 0.f, 0.f, 0.f, 0.f, 0.f, 0.f, 0.f};
  return z;
}

#define WMMA_BF16(A, B, C) \
  __builtin_amdgcn_wmma_f32_16x16x32_bf16(false, (A), false, (B), (short)0, (C), false, false)

// Swin shift-mask region id on the 56-wide patch grid
__device__ __forceinline__ int reg56(int y) { return (y < 49) ? 0 : ((y < 53) ? 1 : 2); }

// ---------------- kernel A: fp32 -> bf16 weights ----------------
__global__ __launch_bounds__(256) void cvt_weights_kernel(
    const float* __restrict__ Wq, const float* __restrict__ Wk,
    const float* __restrict__ Wv, const float* __restrict__ Wo,
    bf16* __restrict__ q, bf16* __restrict__ k,
    bf16* __restrict__ v, bf16* __restrict__ o) {
  int i = blockIdx.x * 256 + threadIdx.x;
  if (i >= WCNT) return;
  q[i] = (bf16)Wq[i];
  k[i] = (bf16)Wk[i];
  v[i] = (bf16)Wv[i];
  o[i] = (bf16)Wo[i];
}

// ---------------- kernel B: pixel-roll(+3,+3) re-patchify -> token-major bf16 ----------------
// 1 block = (n, patch-row gy). Stage 4 wrapped pixel rows x 16 ch in LDS (coalesced reads),
// then emit the 56-token strip [56 x 256] as one contiguous coalesced write.
__global__ __launch_bounds__(256) void shift_tokens_kernel(
    const float* __restrict__ in, bf16* __restrict__ xsG) {
  __shared__ bf16 pix[16 * 4 * 224];   // [c][py][X], 28,672 B
  int n   = blockIdx.x / CH;
  int gy  = blockIdx.x % CH;
  int tid = threadIdx.x;

  // read: 16 ch x 4 rows x 4 px-phases x 56 contiguous floats
  for (int e = tid; e < 16 * 4 * 224; e += 256) {
    int gx2 = e % 56;          // fast index -> contiguous global reads
    int seg = e / 56;
    int px2 = seg & 3;
    int py  = (seg >> 2) & 3;
    int c   = seg >> 4;
    int Y = gy * 4 + py - 3; if (Y < 0) Y += 224;   // rolled[Y] = orig[Y-3 mod 224]
    int d2 = (c << 4) + ((Y & 3) << 2) + px2;
    float v = in[((size_t)n * D_DIM + d2) * PN + (Y >> 2) * CH + gx2];
    pix[((c << 2) + py) * 224 + (gx2 << 2) + px2] = (bf16)v;
  }
  __syncthreads();

  // write: token strip is fully contiguous in xsG
  bf16* dst = xsG + ((size_t)n * PN + (size_t)gy * CH) * D_DIM;
  for (int e = tid; e < CH * D_DIM; e += 256) {
    int d  = e & 255;
    int gx = e >> 8;
    int c = d >> 4, py = (d >> 2) & 3, px = d & 3;
    int X = gx * 4 + px - 3; if (X < 0) X += 224;
    dst[e] = pix[((c << 2) + py) * 224 + X];
  }
}

// ---------------- fused window attention: 1 block = (n, window), 8 wave32 ----------------
__global__ __launch_bounds__(256) void swin_attn_kernel(
    const bf16* __restrict__ xsG,
    const bf16* __restrict__ WqB, const bf16* __restrict__ WkB,
    const bf16* __restrict__ WvB, const bf16* __restrict__ WoB,
    const float* __restrict__ bq, const float* __restrict__ bk,
    const float* __restrict__ bv, const float* __restrict__ bo,
    const float* __restrict__ rel_code, float* __restrict__ out) {
  extern __shared__ char smem[];
  bf16*  xs   = (bf16*)(smem + L_XS);    // [64 tok][XS_S] window tokens
  bf16*  qs   = (bf16*)(smem + L_QS);    // [64 tok][XS_S] Q
  bf16*  ks   = (bf16*)(smem + L_KS);    // [64 tok][XS_S] K  (reused as headT)
  bf16*  vt   = (bf16*)(smem + L_VT);    // [256 d][VT_S]  V transposed
  float* attT = (float*)(smem + L_ATT);  // [64 q][AT_S]   scores, transposed
  bf16*  wmT  = (bf16*)(smem + L_WM);    // [64 q][WM_S]   probs, transposed

  const int tid  = threadIdx.x;
  const int lane = tid & 31;
  const int w    = tid >> 5;
  const int hi8  = (lane >> 4) << 3;

  const int n  = blockIdx.x >> 6;
  const int g  = blockIdx.x & 63;
  const int wy = g >> 3, wx = g & 7;

  // keep weights warm (global_prefetch_b8 path)
  __builtin_prefetch(WqB, 0, 1);
  __builtin_prefetch(WkB, 0, 1);
  __builtin_prefetch(WvB, 0, 1);
  __builtin_prefetch(WoB, 0, 1);

  // stage 0: gather this window's 49 token rows (512B each) into LDS, zero-pad to 64
  {
    unsigned* xw = (unsigned*)xs;
    for (int e = tid; e < PT * (D_DIM / 2); e += 256) {
      int row = e >> 7, c2 = e & 127;
      unsigned val = 0u;
      if (row < P_TOK) {
        int gp = (wy * WSZ + row / WSZ) * CH + wx * WSZ + row % WSZ;
        const unsigned* src = (const unsigned*)(xsG + ((size_t)n * PN + gp) * D_DIM);
        val = src[c2];
      }
      xw[row * (XS_S / 2) + c2] = val;
    }
  }
  __syncthreads();

  // persistent output-projection accumulators:
  // wave w owns output tiles (tt = w>>1, ot = (w&1)*8 + i), i = 0..7
  const int tt_o = w >> 1;
  const int ot0  = (w & 1) * 8;
  v8f facc[8];
#pragma unroll
  for (int i = 0; i < 8; ++i) facc[i] = v8f_zero();

  for (int h = 0; h < HEADS; ++h) {
    // ---- Q/K/V projections: each wave owns 6 weight columns (mat,ft) x 4 token tiles,
    //      B-fragment loaded once per kk and reused across the 4 token tiles ----
    for (int c = 0; c < 6; ++c) {
      int cidx = w + 8 * c;            // 0..47 over (3 mats x 16 feature tiles)
      int mat = cidx >> 4, ft = cidx & 15;
      const bf16* Wb; const float* bb;
      if (mat == 0)      { Wb = WqB; bb = bq; }
      else if (mat == 1) { Wb = WkB; bb = bk; }
      else               { Wb = WvB; bb = bv; }
      Wb += (size_t)h * 256 * 256;     // head slice of [2048 x 256]
      v8f acc[4];
#pragma unroll
      for (int t = 0; t < 4; ++t) acc[t] = v8f_zero();
#pragma unroll
      for (int kk = 0; kk < 8; ++kk) {
        v16bf b = load_b_cm(Wb, 256, kk * 32, ft * 16, lane);
#pragma unroll
        for (int t = 0; t < 4; ++t) {
          v16bf a = load_a_rm(xs, XS_S, t * 16, kk * 32, lane);
          acc[t] = WMMA_BF16(a, b, acc[t]);
        }
      }
      int col = ft * 16 + (lane & 15);
      float bias = bb[h * 256 + col];
      if (mat == 2) {
        // V stored transposed: vt[d][p] -> 8-consecutive-token runs per lane
#pragma unroll
        for (int t = 0; t < 4; ++t)
#pragma unroll
          for (int r = 0; r < 8; ++r)
            vt[col * VT_S + t * 16 + hi8 + r] = (bf16)(acc[t][r] + bias);
      } else {
        bf16* dst = (mat == 0) ? qs : ks;
#pragma unroll
        for (int t = 0; t < 4; ++t)
#pragma unroll
          for (int r = 0; r < 8; ++r)
            dst[(t * 16 + r + hi8) * XS_S + col] = (bf16)(acc[t][r] + bias);
      }
    }
    __syncthreads();

    // ---- scores: attT[q][p] = (sum_d K[p,d] Q[q,d]) / 16 + shift_mask + rel_bias ----
    for (int idx = w; idx < 16; idx += 8) {
      int pt = idx >> 2, qt = idx & 3;
      v8f acc = v8f_zero();
#pragma unroll
      for (int kk = 0; kk < 8; ++kk) {
        v16bf a = load_a_rm(ks, XS_S, pt * 16, kk * 32, lane);
        v16bf b = load_b_cm(qs, XS_S, kk * 32, qt * 16, lane);
        acc = WMMA_BF16(a, b, acc);
      }
      int q = qt * 16 + (lane & 15);
#pragma unroll
      for (int r = 0; r < 8; ++r) {
        int p = pt * 16 + r + hi8;
        float val = -1e30f;
        if (p < P_TOK && q < P_TOK) {
          int rp = p / 7, cp = p % 7, rq = q / 7, cq = q % 7;
          int fmp = reg56(wy * 7 + rp) * 3 + reg56(wx * 7 + cp);
          int fmq = reg56(wy * 7 + rq) * 3 + reg56(wx * 7 + cq);
          float msk = (fmp != fmq) ? -100.f : 0.f;
          int ridx = (rp - rq + 6) + 13 * (cp - cq + 6);
          val = acc[r] * 0.0625f + msk + rel_code[ridx * HEADS + h];
        }
        attT[q * AT_S + p] = val;   // 8 consecutive p per lane -> mergeable stores
      }
    }
    __syncthreads();

    // ---- softmax over keys p: thread q scans its contiguous row ----
    if (tid < 64) {
      int q = tid;
      const float* arow = attT + q * AT_S;
      bf16* wrow = wmT + q * WM_S;
      float mx = -1e30f;
      for (int p = 0; p < P_TOK; ++p) mx = fmaxf(mx, arow[p]);
      float s = 0.f;
      for (int p = 0; p < P_TOK; ++p) s += __expf(arow[p] - mx);
      float inv = 1.f / s;
      for (int p = 0; p < PT; ++p)
        wrow[p] = (p < P_TOK) ? (bf16)(__expf(arow[p] - mx) * inv) : (bf16)0.f;
    }
    __syncthreads();

    // ---- head output: headT[q][d] = sum_p wmT[q][p] * vt[d][p]
    //      wave owns (qt = w>>1, dt = (w&1)*8 + i); A-fragment hoisted across the 8 dt tiles ----
    bf16* hs = ks;  // K no longer needed this head
    {
      v8f oacc[8];
#pragma unroll
      for (int i = 0; i < 8; ++i) oacc[i] = v8f_zero();
#pragma unroll
      for (int kk = 0; kk < 2; ++kk) {
        v16bf a = load_a_rm(wmT, WM_S, tt_o * 16, kk * 32, lane);   // A[m=q][k=p]
#pragma unroll
        for (int i = 0; i < 8; ++i) {
          v16bf b = load_b_cm(vt, VT_S, kk * 32, (ot0 + i) * 16, lane);  // B[k=p][n=d]
          oacc[i] = WMMA_BF16(a, b, oacc[i]);
        }
      }
#pragma unroll
      for (int i = 0; i < 8; ++i) {
        int col = (ot0 + i) * 16 + (lane & 15);
#pragma unroll
        for (int r = 0; r < 8; ++r)
          hs[(tt_o * 16 + r + hi8) * XS_S + col] = (bf16)oacc[i][r];
      }
    }
    __syncthreads();

    // ---- accumulate output projection: y[t][o] += headT(64x256) @ Wo[:, h*256:+256]^T
    //      A-fragment (headT rows) hoisted across the wave's 8 ot tiles ----
#pragma unroll
    for (int kk = 0; kk < 8; ++kk) {
      v16bf a = load_a_rm(hs, XS_S, tt_o * 16, kk * 32, lane);
#pragma unroll
      for (int i = 0; i < 8; ++i) {
        v16bf b = load_b_cm(WoB + h * 256, 2048, kk * 32, (ot0 + i) * 16, lane); // Wo[o][h*256+d]
        facc[i] = WMMA_BF16(a, b, facc[i]);
      }
    }
    __syncthreads();  // protect hs/qs/ks/vt before next head overwrites
  }

  // ---- epilogue: stage [64 tok][256 o] f32 in LDS (reuse xs+qs), then
  //      write +bo with 7-float contiguous runs per (o, window-row) ----
  float* yf = (float*)smem;   // 65,536 B over xs+qs (dead now)
#pragma unroll
  for (int i = 0; i < 8; ++i) {
    int col = (ot0 + i) * 16 + (lane & 15);
#pragma unroll
    for (int r = 0; r < 8; ++r)
      yf[(tt_o * 16 + r + hi8) * 256 + col] = facc[i][r];
  }
  __syncthreads();
  for (int e = tid; e < D_DIM * P_TOK; e += 256) {
    int o = e / P_TOK;
    int t = e - o * P_TOK;
    int gp = (wy * WSZ + t / WSZ) * CH + wx * WSZ + t % WSZ;
    out[((size_t)n * D_DIM + o) * PN + gp] = yf[t * 256 + o] + bo[o];
  }
}

// ---------------- host launch ----------------
extern "C" void kernel_launch(void* const* d_in, const int* in_sizes, int n_in,
                              void* d_out, int out_size, void* d_ws, size_t ws_size,
                              hipStream_t stream) {
  const float* inputs = (const float*)d_in[0];
  const float* Wk     = (const float*)d_in[1];
  const float* bk     = (const float*)d_in[2];
  const float* Wq     = (const float*)d_in[3];
  const float* bq     = (const float*)d_in[4];
  const float* Wv     = (const float*)d_in[5];
  const float* bv     = (const float*)d_in[6];
  const float* Wo     = (const float*)d_in[7];
  const float* bo     = (const float*)d_in[8];
  const float* rel    = (const float*)d_in[9];
  float* out = (float*)d_out;

  char* ws = (char*)d_ws;
  bf16* xsG = (bf16*)ws;
  bf16* WqB = (bf16*)(ws + XS_BYTES);
  bf16* WkB = (bf16*)(ws + XS_BYTES + W_BYTES);
  bf16* WvB = (bf16*)(ws + XS_BYTES + 2 * W_BYTES);
  bf16* WoB = (bf16*)(ws + XS_BYTES + 3 * W_BYTES);

  cvt_weights_kernel<<<(WCNT + 255) / 256, 256, 0, stream>>>(Wq, Wk, Wv, Wo, WqB, WkB, WvB, WoB);

  shift_tokens_kernel<<<N_BATCH * CH, 256, 0, stream>>>(inputs, xsG);

  swin_attn_kernel<<<N_BATCH * NWIN, 256, L_TOT, stream>>>(
      xsG, WqB, WkB, WvB, WoB, bq, bk, bv, bo, rel, out);
}
